// TiledAttention_6691559047431
// MI455X (gfx1250) — compile-verified, matrix operand order checked by reading, not run on hardware
//
#include <hip/hip_runtime.h>
#include <cstdint>

// ---------------------------------------------------------------------------
// MHA forward for MI455X (gfx1250, wave32, WMMA + Tensor Data Mover).
//   hidden [2,2048,1024] fp32 -> QKV proj (bf16 out) -> flash attn -> O proj.
// Matmuls: v_wmma_f32_16x16x32_bf16 (bf16 operands, f32 accumulate).
// Global->LDS staging: TDM (tensor_load_to_lds), double-buffered, D# pad
// fields generate bank-conflict-free LDS row strides in hardware.
// Workspace: Q|K|V bf16 (8 MB each) + CTX fp32 (16 MB) = 40 MB.
// ---------------------------------------------------------------------------

typedef __attribute__((ext_vector_type(16))) __bf16        v16bf;
typedef __attribute__((ext_vector_type(8)))  __bf16        v8bf;
typedef __attribute__((ext_vector_type(8)))  float         v8f;
typedef __attribute__((ext_vector_type(4)))  unsigned int  v4u;
typedef __attribute__((ext_vector_type(8)))  int           v8i;
typedef __attribute__((ext_vector_type(4)))  int           v4i;

#define HSZ   1024
#define SEQ   2048
#define NHEAD 16
#define HDIM  64

__device__ __forceinline__ __bf16 f2bf(float f) { return (__bf16)f; }

__device__ __forceinline__ v8f wmma_bf16(v16bf a, v16bf b, v8f c) {
  return __builtin_amdgcn_wmma_f32_16x16x32_bf16(
      false, a, false, b, (short)0, c, false, false);
}

// Fragment from fp32 memory (LDS): elems 0..7 = base[0..7], 8..15 = base[16..23]
__device__ __forceinline__ v16bf load_frag_f32(const float* base) {
  float4 a0 = *(const float4*)(base);
  float4 a1 = *(const float4*)(base + 4);
  float4 b0 = *(const float4*)(base + 16);
  float4 b1 = *(const float4*)(base + 20);
  v16bf r;
  r[0]  = f2bf(a0.x); r[1]  = f2bf(a0.y); r[2]  = f2bf(a0.z); r[3]  = f2bf(a0.w);
  r[4]  = f2bf(a1.x); r[5]  = f2bf(a1.y); r[6]  = f2bf(a1.z); r[7]  = f2bf(a1.w);
  r[8]  = f2bf(b0.x); r[9]  = f2bf(b0.y); r[10] = f2bf(b0.z); r[11] = f2bf(b0.w);
  r[12] = f2bf(b1.x); r[13] = f2bf(b1.y); r[14] = f2bf(b1.z); r[15] = f2bf(b1.w);
  return r;
}

// Fragment from bf16 memory (global or LDS; b128 loads, no conversion):
// elems 0..7 = p[0..7], 8..15 = p[16..23]
__device__ __forceinline__ v16bf load_frag_bf16(const __bf16* p) {
  v8bf lo = *(const v8bf*)(p);
  v8bf hi = *(const v8bf*)(p + 16);
  v16bf r;
#pragma unroll
  for (int i = 0; i < 8; ++i) { r[i] = lo[i]; r[8 + i] = hi[i]; }
  return r;
}

// ---------------- Tensor Data Mover descriptor helpers ---------------------
// D# group0: count=1 | lds_addr | global_addr | type=2 ("image")
__device__ __forceinline__ v4u tdm_group0(unsigned lds_byte_addr, const void* gptr) {
  unsigned long long ga = (unsigned long long)(size_t)gptr;
  v4u g;
  g[0] = 1u;
  g[1] = lds_byte_addr;
  g[2] = (unsigned)ga;
  g[3] = ((unsigned)(ga >> 32) & 0x01FFFFFFu) | (2u << 30);
  return g;
}

// D# group1. ds_code: 0=1B,1=2B,2=4B. Pad: interval 2^(pad_code+1) dwords,
// amount 4 dwords (code 3) appended per tile row.
__device__ __forceinline__ v8i tdm_group1(unsigned ds_code, unsigned tile_d0,
                                          unsigned tile_rows, unsigned tensor_d0,
                                          unsigned stride_elems, unsigned pad_code) {
  const unsigned td1 = 1u << 20;  // generous row count for OOB check
  v8i g;
  g[0] = (int)((ds_code << 16) | (1u << 20) | (pad_code << 22) | (3u << 25));
  g[1] = (int)((tensor_d0 & 0xFFFFu) << 16);
  g[2] = (int)(((tensor_d0 >> 16) & 0xFFFFu) | ((td1 & 0xFFFFu) << 16));
  g[3] = (int)(((td1 >> 16) & 0xFFFFu) | (tile_d0 << 16));
  g[4] = (int)tile_rows;
  g[5] = (int)stride_elems;
  g[6] = 0;
  g[7] = 0;
  return g;
}

__device__ __forceinline__ void tdm_load(const void* gsrc, unsigned lds_byte_addr,
                                         unsigned ds_code, unsigned tile_d0,
                                         unsigned tile_rows, unsigned tensor_d0,
                                         unsigned stride_elems, unsigned pad_code) {
  const v4i z4 = {0, 0, 0, 0};
  const v8i z8 = {0, 0, 0, 0, 0, 0, 0, 0};
  __builtin_amdgcn_tensor_load_to_lds(tdm_group0(lds_byte_addr, gsrc),
                                      tdm_group1(ds_code, tile_d0, tile_rows,
                                                 tensor_d0, stride_elems, pad_code),
                                      z4, z4, z8, 0);
}

// ---------------------------------------------------------------------------
// GEMM: out = X[4096,1024] @ W[1024,1024]^T + bias, then * out_scale.
// QKV_LAYOUT=1: bf16 output scattered to [B,NH,S,HD];  0: fp32 [4096,1024].
// Block = 256 thr (8 waves), WG tile 64x128, wave tile 32x32 (2x2 WMMA).
// A/B fp32 tiles DMA'd by TDM into double-buffered LDS (row stride 36 dw).
// ---------------------------------------------------------------------------
template <int QKV_LAYOUT>
__global__ __launch_bounds__(256)
void gemm_wmma_kernel(const float* __restrict__ X, const float* __restrict__ W,
                      const float* __restrict__ bias, void* __restrict__ outv,
                      float out_scale) {
  __shared__ float Asf[2][64 * 36];
  __shared__ float Bsf[2][128 * 36];

  const int t     = threadIdx.x;
  const int tileM = blockIdx.x >> 3;
  const int tileN = blockIdx.x & 7;
  const int w     = t >> 5;
  const int lane  = t & 31;
  const int half  = lane >> 4;
  const int l15   = lane & 15;
  const int koff  = 8 * half;
  const int waveM = w >> 2;
  const int waveN = w & 3;

  const float* Abase = X + (size_t)(tileM * 64) * HSZ;
  const float* Bbase = W + (size_t)(tileN * 128) * HSZ;
  const unsigned ldsA[2] = {(unsigned)(size_t)&Asf[0][0], (unsigned)(size_t)&Asf[1][0]};
  const unsigned ldsB[2] = {(unsigned)(size_t)&Bsf[0][0], (unsigned)(size_t)&Bsf[1][0]};

  v8f acc[2][2];
#pragma unroll
  for (int mi = 0; mi < 2; ++mi)
#pragma unroll
    for (int ni = 0; ni < 2; ++ni) acc[mi][ni] = 0.0f;

  // pre-issue kk = 0 (fp32, 32-dword rows -> pad_code 4, LDS stride 36 dw)
  if (w == 0)      tdm_load(Abase, ldsA[0], 2, 32, 64,  HSZ, HSZ, 4);
  else if (w == 1) tdm_load(Bbase, ldsB[0], 2, 32, 128, HSZ, HSZ, 4);

  for (int kk = 0; kk < HSZ; kk += 32) {
    const int par = (kk >> 5) & 1;
    if (kk + 32 < HSZ) {
      if (w == 0)      tdm_load(Abase + kk + 32, ldsA[1 - par], 2, 32, 64,  HSZ, HSZ, 4);
      else if (w == 1) tdm_load(Bbase + kk + 32, ldsB[1 - par], 2, 32, 128, HSZ, HSZ, 4);
      if (w < 2) __builtin_amdgcn_s_wait_tensorcnt(1);
    } else {
      if (w < 2) __builtin_amdgcn_s_wait_tensorcnt(0);
    }
    __syncthreads();

    const float* Ab = &Asf[par][0];
    const float* Bb = &Bsf[par][0];
    v16bf aF[2], bF[2];
#pragma unroll
    for (int mi = 0; mi < 2; ++mi)
      aF[mi] = load_frag_f32(Ab + (waveM * 32 + mi * 16 + l15) * 36 + koff);
#pragma unroll
    for (int ni = 0; ni < 2; ++ni)
      bF[ni] = load_frag_f32(Bb + (waveN * 32 + ni * 16 + l15) * 36 + koff);
#pragma unroll
    for (int mi = 0; mi < 2; ++mi)
#pragma unroll
      for (int ni = 0; ni < 2; ++ni)
        acc[mi][ni] = wmma_bf16(aF[mi], bF[ni], acc[mi][ni]);

    __syncthreads();
  }

  // epilogue: VGPR g holds row g+8*half, col = l15
#pragma unroll
  for (int mi = 0; mi < 2; ++mi) {
#pragma unroll
    for (int ni = 0; ni < 2; ++ni) {
      const int ncol = tileN * 128 + waveN * 32 + ni * 16 + l15;
      const float bv = bias[ncol];
#pragma unroll
      for (int g = 0; g < 8; ++g) {
        const int mrow = tileM * 64 + waveM * 32 + mi * 16 + g + 8 * half;
        const float val = (acc[mi][ni][g] + bv) * out_scale;
        if (QKV_LAYOUT) {
          const int b = mrow >> 11, s = mrow & (SEQ - 1);
          const int h = ncol >> 6, d = ncol & (HDIM - 1);
          ((__bf16*)outv)[(((size_t)(b * NHEAD + h) * SEQ) + s) * HDIM + d] = f2bf(val);
        } else {
          ((float*)outv)[(size_t)mrow * HSZ + ncol] = val;
        }
      }
    }
  }
}

// ---------------------------------------------------------------------------
// Flash attention over bf16 Q/K/V (scale pre-folded into Q by the GEMM).
// Block = 256 thr (8 waves); wave owns 16 query rows x full HD=64.
// K/V 32x64 bf16 tiles staged per block via double-buffered TDM
// (2B elems, 128B rows -> pad_code 4 -> LDS stride 72 bf16 = 36 dw).
// grid = B*NH * (S/128) = 512 blocks.
// ---------------------------------------------------------------------------
__global__ __launch_bounds__(256)
void flash_wmma_kernel(const __bf16* __restrict__ Q, const __bf16* __restrict__ Kg,
                       const __bf16* __restrict__ Vg, float* __restrict__ CTX) {
  __shared__ __bf16 Ksh[2][32 * 72];
  __shared__ __bf16 Vsh[2][32 * 72];
  __shared__ __bf16 Psh[8 * 16 * 40];  // per-wave 16x32 bf16 P tile

  const int bh   = blockIdx.x >> 4;
  const int qblk = blockIdx.x & 15;
  const int t    = threadIdx.x;
  const int w    = t >> 5;
  const int lane = t & 31;
  const int half = lane >> 4;
  const int l15  = lane & 15;
  const int koff = 8 * half;
  const int q0   = qblk * 128 + w * 16;

  const __bf16* Qh = Q  + (size_t)bh * SEQ * HDIM;
  const __bf16* Kh = Kg + (size_t)bh * SEQ * HDIM;
  const __bf16* Vh = Vg + (size_t)bh * SEQ * HDIM;
  __bf16* Pb = Psh + w * (16 * 40);
  const unsigned ldsK[2] = {(unsigned)(size_t)&Ksh[0][0], (unsigned)(size_t)&Ksh[1][0]};
  const unsigned ldsV[2] = {(unsigned)(size_t)&Vsh[0][0], (unsigned)(size_t)&Vsh[1][0]};

  // Q fragments straight from global bf16 (global_load_b128, no cvt)
  v16bf aq[2];
#pragma unroll
  for (int c = 0; c < 2; ++c)
    aq[c] = load_frag_bf16(Qh + (size_t)(q0 + l15) * HDIM + c * 32 + koff);

  v8f accO[4];
#pragma unroll
  for (int j = 0; j < 4; ++j) accO[j] = 0.0f;
  float m_i[8], l_i[8];
#pragma unroll
  for (int g = 0; g < 8; ++g) { m_i[g] = -1e30f; l_i[g] = 0.0f; }

  // pre-issue n0 = 0 (bf16: ds_code 1, 128B rows -> pad_code 4)
  if (w == 0)      tdm_load(Kh, ldsK[0], 1, HDIM, 32, HDIM, HDIM, 4);
  else if (w == 1) tdm_load(Vh, ldsV[0], 1, HDIM, 32, HDIM, HDIM, 4);

  for (int n0 = 0; n0 < SEQ; n0 += 32) {
    const int par = (n0 >> 5) & 1;
    if (n0 + 32 < SEQ) {
      if (w == 0)      tdm_load(Kh + (size_t)(n0 + 32) * HDIM, ldsK[1 - par], 1, HDIM, 32, HDIM, HDIM, 4);
      else if (w == 1) tdm_load(Vh + (size_t)(n0 + 32) * HDIM, ldsV[1 - par], 1, HDIM, 32, HDIM, HDIM, 4);
      if (w < 2) __builtin_amdgcn_s_wait_tensorcnt(1);
    } else {
      if (w < 2) __builtin_amdgcn_s_wait_tensorcnt(0);
    }
    __syncthreads();

    const __bf16* Kb = &Ksh[par][0];
    const __bf16* Vb = &Vsh[par][0];

    // scores: two 16x16 tiles over HD=64 (2 k-chunks each); pure ds_load_b128
    v8f sA = 0.0f, sB = 0.0f;
    {
      v16bf bk;
      bk = load_frag_bf16(Kb + (l15)*72 + koff);            sA = wmma_bf16(aq[0], bk, sA);
      bk = load_frag_bf16(Kb + (l15)*72 + 32 + koff);       sA = wmma_bf16(aq[1], bk, sA);
      bk = load_frag_bf16(Kb + (16 + l15)*72 + koff);       sB = wmma_bf16(aq[0], bk, sB);
      bk = load_frag_bf16(Kb + (16 + l15)*72 + 32 + koff);  sB = wmma_bf16(aq[1], bk, sB);
    }

    // online softmax; row g lives in this half-wave's 16 lanes
#pragma unroll
    for (int g = 0; g < 8; ++g) {
      float s0 = sA[g], s1 = sB[g];
      float mx = fmaxf(s0, s1);
      mx = fmaxf(mx, __shfl_xor(mx, 1, 32));
      mx = fmaxf(mx, __shfl_xor(mx, 2, 32));
      mx = fmaxf(mx, __shfl_xor(mx, 4, 32));
      mx = fmaxf(mx, __shfl_xor(mx, 8, 32));
      const float mnew  = fmaxf(m_i[g], mx);
      const float alpha = __expf(m_i[g] - mnew);
      const float p0 = __expf(s0 - mnew);
      const float p1 = __expf(s1 - mnew);
      float rs = p0 + p1;
      rs += __shfl_xor(rs, 1, 32);
      rs += __shfl_xor(rs, 2, 32);
      rs += __shfl_xor(rs, 4, 32);
      rs += __shfl_xor(rs, 8, 32);
      l_i[g] = l_i[g] * alpha + rs;
      m_i[g] = mnew;
      accO[0][g] *= alpha; accO[1][g] *= alpha;
      accO[2][g] *= alpha; accO[3][g] *= alpha;
      const int prow = g + 8 * half;
      Pb[prow * 40 + l15]      = f2bf(p0);
      Pb[prow * 40 + 16 + l15] = f2bf(p1);
    }

    // wave-local C-layout -> A-fragment conversion through LDS
    asm volatile("s_wait_dscnt 0" ::: "memory");
    v16bf ap = load_frag_bf16(Pb + l15 * 40 + koff);

    // O += P(16x32) @ V(32x64): bare 16-bit LDS gathers, no cvt
#pragma unroll
    for (int j = 0; j < 4; ++j) {
      v16bf bv;
      const int d = j * 16 + l15;
#pragma unroll
      for (int i = 0; i < 8; ++i) {
        bv[i]     = Vb[(koff + i) * 72 + d];
        bv[8 + i] = Vb[(16 + koff + i) * 72 + d];
      }
      accO[j] = wmma_bf16(ap, bv, accO[j]);
    }

    __syncthreads();  // buf[par] free before next TDM refill
  }

  // normalize + write CTX [B,S,H] fp32
  const int b = bh >> 4, h = bh & (NHEAD - 1);
#pragma unroll
  for (int j = 0; j < 4; ++j) {
    const int d = h * HDIM + j * 16 + l15;
#pragma unroll
    for (int g = 0; g < 8; ++g) {
      const int srow = q0 + g + 8 * half;
      CTX[((size_t)b * SEQ + srow) * HSZ + d] = accO[j][g] / l_i[g];
    }
  }
}

// ---------------------------------------------------------------------------
extern "C" void kernel_launch(void* const* d_in, const int* in_sizes, int n_in,
                              void* d_out, int out_size, void* d_ws, size_t ws_size,
                              hipStream_t stream) {
  (void)in_sizes; (void)n_in; (void)out_size; (void)ws_size;
  const float* hs = (const float*)d_in[0];
  const float* Wq = (const float*)d_in[1];
  const float* bq = (const float*)d_in[2];
  const float* Wk = (const float*)d_in[3];
  const float* bk = (const float*)d_in[4];
  const float* Wv = (const float*)d_in[5];
  const float* bv = (const float*)d_in[6];
  const float* Wo = (const float*)d_in[7];
  const float* bo = (const float*)d_in[8];

  const size_t perTensor = (size_t)2 * NHEAD * SEQ * HDIM;  // 4,194,304 elems
  __bf16* Qd   = (__bf16*)d_ws;
  __bf16* Kd   = Qd + perTensor;
  __bf16* Vd   = Kd + perTensor;
  float*  CTXd = (float*)(Vd + perTensor);

  dim3 block(256);
  dim3 gridG(512);   // (4096/64) * (1024/128)
  dim3 gridF(512);   // 32 (b,h) * (2048/128)

  const float qscale = 0.125f;  // 1/sqrt(64) folded into Q
  gemm_wmma_kernel<1><<<gridG, block, 0, stream>>>(hs, Wq, bq, Qd, qscale);
  gemm_wmma_kernel<1><<<gridG, block, 0, stream>>>(hs, Wk, bk, Kd, 1.0f);
  gemm_wmma_kernel<1><<<gridG, block, 0, stream>>>(hs, Wv, bv, Vd, 1.0f);
  flash_wmma_kernel<<<gridF, block, 0, stream>>>(Qd, Kd, Vd, CTXd);
  gemm_wmma_kernel<0><<<gridG, block, 0, stream>>>(CTXd, Wo, bo, d_out, 1.0f);
}
